// GraphReasoningLayer_24713241821558
// MI455X (gfx1250) — matrix operand order repaired and checked by reading, not compiled, not run on hardware
//
#include <hip/hip_runtime.h>

// ---------------------------------------------------------------------------
// CDNA5 (gfx1250) GraphReasoningLayer.
// All dense GEMMs via v_wmma_f32_16x16x32_f16 with fp32 accumulate.
// GEMM operands are pre-converted to f16 ONCE (weights at start, activations
// per use; LN-ReLU and softmax emit f16 directly), so the GEMM hot loop is
// pure b128 loads + wmma with no conversion VALU.
// wave32; 4 waves / 128-thread block; wave tile 32(M) x 64(N) = 8 wmma/K-step;
// K-loop unrolled x2 so fragment loads double-buffer (avoids WMMA WAR nops).
// ---------------------------------------------------------------------------

typedef _Float16 f16;
typedef _Float16 v4h  __attribute__((ext_vector_type(4)));
typedef _Float16 v8h  __attribute__((ext_vector_type(8)));
typedef _Float16 v16h __attribute__((ext_vector_type(16)));
typedef float    v8f  __attribute__((ext_vector_type(8)));

// A fragment (16x32, row-major [M,K] f16): lane half h holds K=16h..16h+15
__device__ __forceinline__ v16h load_a_frag16(const f16* __restrict__ A, int lda,
                                              int row, int k0, int lane) {
  const int half = lane >> 4;
  const f16* p = A + (size_t)row * lda + k0 + 16 * half;
  const v8h lo = *reinterpret_cast<const v8h*>(p);       // 16B aligned
  const v8h hi = *reinterpret_cast<const v8h*>(p + 8);   // 16B aligned
  v16h r;
#pragma unroll
  for (int i = 0; i < 8; ++i) { r[i] = lo[i]; r[8 + i] = hi[i]; }
  return r;
}

// B fragment from row-major [N,K] f16 (torch weight layout): lane half h holds K=16h..16h+15
__device__ __forceinline__ v16h load_b_frag16_nk(const f16* __restrict__ B, int ldb,
                                                 int col, int k0, int lane) {
  const int half = lane >> 4;
  const f16* p = B + (size_t)col * ldb + k0 + 16 * half;
  const v8h lo = *reinterpret_cast<const v8h*>(p);
  const v8h hi = *reinterpret_cast<const v8h*>(p + 8);
  v16h r;
#pragma unroll
  for (int i = 0; i < 8; ++i) { r[i] = lo[i]; r[8 + i] = hi[i]; }
  return r;
}

// B fragment from row-major [K,N] f16 (used for P @ V): strided gather
__device__ __forceinline__ v16h load_b_frag16_kn(const f16* __restrict__ B, int ldb,
                                                 int col, int k0, int lane) {
  const int half = lane >> 4;
  const f16* p = B + (size_t)(k0 + 16 * half) * ldb + col;
  v16h r;
#pragma unroll
  for (int i = 0; i < 16; ++i) r[i] = p[(size_t)i * ldb];
  return r;
}

// C[m,n] = alpha * sum_k A[m,k]*B(k,n)  (+ bias[n]) (+ residual[m,n]);  C is fp32.
// Block tile 128(M) x 64(N), 4 waves; each wave 32(M) x 64(N), 8 accumulators.
// Branch-free K-loop: load columns clamped in-bounds, wmmas unconditional,
// stores guarded by wave-uniform tile-validity tests (EXEC all-ones at wmma).
template <bool B_IS_KxN>
__global__ __launch_bounds__(128)
void gemm_wmma_kernel(const f16* __restrict__ A, int lda,
                      const f16* __restrict__ B, int ldb,
                      const float* __restrict__ bias,
                      const float* residual, int ldr,
                      float* C, int ldc,
                      int M, int N, int K, float alpha) {
  const int lane = threadIdx.x & 31;
  const int wave = threadIdx.x >> 5;
  const int m0 = blockIdx.y * 128 + wave * 32;
  const int n0 = blockIdx.x * 64;
  if (m0 >= M) return;

  v8f acc0[4], acc1[4];
#pragma unroll
  for (int j = 0; j < 4; ++j)
#pragma unroll
    for (int i = 0; i < 8; ++i) { acc0[j][i] = 0.0f; acc1[j][i] = 0.0f; }

  bool valid[4];   // wave-uniform
  int  col[4];     // clamped so loads stay in-bounds even for invalid tiles
#pragma unroll
  for (int j = 0; j < 4; ++j) {
    valid[j] = (n0 + 16 * j) < N;
    int c = n0 + 16 * j + (lane & 15);
    col[j] = c < N ? c : (N - 1);
  }

  const int r0 = m0 + (lane & 15);
  const int r1 = r0 + 16;

#pragma unroll 2
  for (int k0 = 0; k0 < K; k0 += 32) {
    // speculative prefetch of next K-tile (global_prefetch_b8, near scope);
    // safe past the end (speculative prefetch translation failures are dropped)
    __builtin_prefetch(A + (size_t)r0 * lda + k0 + 32, 0, 3);
    __builtin_prefetch(A + (size_t)r1 * lda + k0 + 32, 0, 3);

    v16h a0 = load_a_frag16(A, lda, r0, k0, lane);
    v16h a1 = load_a_frag16(A, lda, r1, k0, lane);

    v16h b[4];
#pragma unroll
    for (int j = 0; j < 4; ++j) {
      if (B_IS_KxN) {
        b[j] = load_b_frag16_kn(B, ldb, col[j], k0, lane);
        __builtin_prefetch(B + (size_t)(k0 + 32) * ldb + col[j], 0, 3);
      } else {
        b[j] = load_b_frag16_nk(B, ldb, col[j], k0, lane);
        __builtin_prefetch(B + (size_t)col[j] * ldb + k0 + 32, 0, 3);
      }
    }

#pragma unroll
    for (int j = 0; j < 4; ++j)
      acc0[j] = __builtin_amdgcn_wmma_f32_16x16x32_f16(
          false, a0, false, b[j], (short)0, acc0[j], false, false);
#pragma unroll
    for (int j = 0; j < 4; ++j)
      acc1[j] = __builtin_amdgcn_wmma_f32_16x16x32_f16(
          false, a1, false, b[j], (short)0, acc1[j], false, false);
  }

  const int half = lane >> 4;
#pragma unroll
  for (int j = 0; j < 4; ++j) {
    if (!valid[j]) continue;  // wave-uniform
    const int c = n0 + 16 * j + (lane & 15);
    const float bb = bias ? bias[c] : 0.0f;
#pragma unroll
    for (int r = 0; r < 8; ++r) {
      // C/D layout: VGPR r -> M = r + 8*half, N = lane&15
      const int row0 = m0 + r + 8 * half;
      const int row1 = row0 + 16;
      float v0 = acc0[j][r] * alpha + bb;
      float v1 = acc1[j][r] * alpha + bb;
      if (residual) {
        v0 += residual[(size_t)row0 * ldr + c];
        v1 += residual[(size_t)row1 * ldr + c];
      }
      C[(size_t)row0 * ldc + c] = v0;
      C[(size_t)row1 * ldc + c] = v1;
    }
  }
}

// fp32 -> f16 bulk convert; n4 = count/4; float4 in, 4xf16 (8B) out
__global__ __launch_bounds__(256)
void cvt_f32_f16_kernel(const float* __restrict__ x, f16* __restrict__ y, int n4) {
  const int i = blockIdx.x * 256 + threadIdx.x;
  if (i >= n4) return;
  const float4 v = reinterpret_cast<const float4*>(x)[i];
  v4h o;
  o[0] = (f16)v.x; o[1] = (f16)v.y; o[2] = (f16)v.z; o[3] = (f16)v.w;
  reinterpret_cast<v4h*>(y)[i] = o;
}

// msgs = nodes[src] + nodes[dst] + edge_proj ; agg[dst] += msgs   (one block / edge)
__global__ __launch_bounds__(192)
void edge_msg_kernel(const float* __restrict__ nodes,
                     const float* __restrict__ eproj,
                     const int* __restrict__ src,
                     const int* __restrict__ dst,
                     float* agg, int E) {
  const int e = blockIdx.x;
  if (e >= E) return;
  const int s = src[e], d = dst[e];
  const int c = threadIdx.x;  // 0..191 float4 chunks of 768 floats
  const float4 a  = reinterpret_cast<const float4*>(nodes + (size_t)s * 768)[c];
  const float4 b4 = reinterpret_cast<const float4*>(nodes + (size_t)d * 768)[c];
  const float4 g  = reinterpret_cast<const float4*>(eproj + (size_t)e * 768)[c];
  float* out = agg + (size_t)d * 768 + (size_t)c * 4;
  unsafeAtomicAdd(out + 0, a.x + b4.x + g.x);
  unsafeAtomicAdd(out + 1, a.y + b4.y + g.y);
  unsafeAtomicAdd(out + 2, a.z + b4.z + g.z);
  unsafeAtomicAdd(out + 3, a.w + b4.w + g.w);
}

// LayerNorm(x)*g+b then ReLU, row length 768; emits f16 (only consumed as GEMM A)
__global__ __launch_bounds__(256)
void ln_relu_kernel(const float* __restrict__ x, const float* __restrict__ g,
                    const float* __restrict__ b, f16* __restrict__ y) {
  const int row = blockIdx.x;
  const float* xr = x + (size_t)row * 768;
  float v[3], s = 0.0f, ss = 0.0f;
#pragma unroll
  for (int i = 0; i < 3; ++i) {
    v[i] = xr[threadIdx.x + 256 * i];
    s += v[i];
    ss += v[i] * v[i];
  }
  __shared__ float rs[256], rq[256];
  rs[threadIdx.x] = s;
  rq[threadIdx.x] = ss;
  __syncthreads();
  for (int off = 128; off > 0; off >>= 1) {
    if (threadIdx.x < off) {
      rs[threadIdx.x] += rs[threadIdx.x + off];
      rq[threadIdx.x] += rq[threadIdx.x + off];
    }
    __syncthreads();
  }
  const float mu = rs[0] * (1.0f / 768.0f);
  const float var = rq[0] * (1.0f / 768.0f) - mu * mu;
  const float inv = rsqrtf(var + 1e-5f);
  f16* yr = y + (size_t)row * 768;
#pragma unroll
  for (int i = 0; i < 3; ++i) {
    const int c = threadIdx.x + 256 * i;
    const float t = (v[i] - mu) * inv * g[c] + b[c];
    yr[c] = (f16)(t > 0.0f ? t : 0.0f);
  }
}

// Row softmax over fixed row length 4096; fp32 in, f16 probabilities out.
// 256 threads/row, 16 elements per thread held in registers (single global read).
__global__ __launch_bounds__(256)
void softmax4096_f16_kernel(const float* __restrict__ scores, f16* __restrict__ probs) {
  const int row = blockIdx.x;
  const float* r = scores + (size_t)row * 4096;
  f16* o = probs + (size_t)row * 4096;
  float v[16];
  float m = -__builtin_inff();
#pragma unroll
  for (int i = 0; i < 16; ++i) {
    v[i] = r[threadIdx.x + 256 * i];
    m = fmaxf(m, v[i]);
  }
  __shared__ float red[256];
  red[threadIdx.x] = m;
  __syncthreads();
  for (int off = 128; off > 0; off >>= 1) {
    if (threadIdx.x < off) red[threadIdx.x] = fmaxf(red[threadIdx.x], red[threadIdx.x + off]);
    __syncthreads();
  }
  m = red[0];
  __syncthreads();
  float s = 0.0f;
#pragma unroll
  for (int i = 0; i < 16; ++i) {
    v[i] = __expf(v[i] - m);
    s += v[i];
  }
  red[threadIdx.x] = s;
  __syncthreads();
  for (int off = 128; off > 0; off >>= 1) {
    if (threadIdx.x < off) red[threadIdx.x] += red[threadIdx.x + off];
    __syncthreads();
  }
  const float inv = 1.0f / red[0];
#pragma unroll
  for (int i = 0; i < 16; ++i) o[threadIdx.x + 256 * i] = (f16)(v[i] * inv);
}

// ---------------------------------------------------------------------------

static inline void run_gemm_nk(hipStream_t st, const f16* A, int lda,
                               const f16* B, int ldb, const float* bias,
                               const float* res, int ldr, float* C, int ldc,
                               int M, int N, int K, float alpha = 1.0f) {
  dim3 grid((N + 63) / 64, (M + 127) / 128);
  gemm_wmma_kernel<false><<<grid, 128, 0, st>>>(A, lda, B, ldb, bias, res, ldr,
                                                C, ldc, M, N, K, alpha);
}

static inline void run_gemm_kn(hipStream_t st, const f16* A, int lda,
                               const f16* B, int ldb, const float* bias,
                               const float* res, int ldr, float* C, int ldc,
                               int M, int N, int K, float alpha = 1.0f) {
  dim3 grid((N + 63) / 64, (M + 127) / 128);
  gemm_wmma_kernel<true><<<grid, 128, 0, st>>>(A, lda, B, ldb, bias, res, ldr,
                                               C, ldc, M, N, K, alpha);
}

static inline void cvt(hipStream_t st, const float* x, f16* y, size_t n) {
  const int n4 = (int)(n / 4);  // all sizes here are multiples of 4
  cvt_f32_f16_kernel<<<(n4 + 255) / 256, 256, 0, st>>>(x, y, n4);
}

extern "C" void kernel_launch(void* const* d_in, const int* in_sizes, int n_in,
                              void* d_out, int out_size, void* d_ws, size_t ws_size,
                              hipStream_t stream) {
  const float* node_features = (const float*)d_in[0];
  const int*   edge_index    = (const int*)d_in[1];
  const float* edge_features = (const float*)d_in[2];
  const float* Wn   = (const float*)d_in[3];
  const float* bn   = (const float*)d_in[4];
  const float* We   = (const float*)d_in[5];
  const float* be   = (const float*)d_in[6];
  const float* W1   = (const float*)d_in[7];
  const float* b1   = (const float*)d_in[8];
  const float* ln_g = (const float*)d_in[9];
  const float* ln_b = (const float*)d_in[10];
  const float* W2   = (const float*)d_in[11];
  const float* b2   = (const float*)d_in[12];
  const float* Wqkv = (const float*)d_in[13];
  const float* bqkv = (const float*)d_in[14];
  const float* Wo   = (const float*)d_in[15];
  const float* bo   = (const float*)d_in[16];
  float* out = (float*)d_out;

  constexpr int Nn = 4096, E = 131072, D = 768, DE = 256, Lc = 3, H = 8, DH = 96;

  // ---- fp32 workspace region ----
  float* ws = (float*)d_ws;
  size_t off = 0;
  float* nodes  = ws + off; off += (size_t)Nn * D;        // 12 MB
  float* agg    = ws + off; off += (size_t)Nn * D;        // 12 MB
  float* h1     = ws + off; off += (size_t)Nn * D;        // 12 MB
  float* attn   = ws + off; off += (size_t)Nn * D;        // 12 MB
  float* qkv    = ws + off; off += (size_t)Nn * 3 * D;    // 38 MB
  float* scores = ws + off; off += (size_t)Nn * Nn;       // 64 MB (reused per head)
  float* eproj  = ws + off; off += (size_t)E * D;         // 403 MB

  // ---- f16 workspace region (starts 4B-aligned; all counts multiples of 8) ----
  f16* wsh = (f16*)(ws + off);
  size_t hoff = 0;
  f16* nf16   = wsh + hoff; hoff += (size_t)Nn * D;        // node_features
  f16* ef16   = wsh + hoff; hoff += (size_t)E * DE;        // edge_features
  f16* wn16   = wsh + hoff; hoff += (size_t)D * D;
  f16* we16   = wsh + hoff; hoff += (size_t)D * DE;
  f16* w116   = wsh + hoff; hoff += (size_t)Lc * D * 2 * D;
  f16* w216   = wsh + hoff; hoff += (size_t)Lc * D * D;
  f16* wqkv16 = wsh + hoff; hoff += (size_t)3 * D * D;
  f16* wo16   = wsh + hoff; hoff += (size_t)D * D;
  f16* x16    = wsh + hoff; hoff += (size_t)Nn * D;        // activation staging
  f16* agg16  = wsh + hoff; hoff += (size_t)Nn * D;
  f16* h216   = wsh + hoff; hoff += (size_t)Nn * D;        // LN-ReLU output
  f16* qkv16  = wsh + hoff; hoff += (size_t)Nn * 3 * D;
  f16* p16    = wsh + hoff; hoff += (size_t)Nn * Nn;       // softmax probs

  const int* src = edge_index;
  const int* dst = edge_index + E;

  // one-time f16 staging of inputs and all weights
  cvt(stream, node_features, nf16, (size_t)Nn * D);
  cvt(stream, edge_features, ef16, (size_t)E * DE);
  cvt(stream, Wn, wn16, (size_t)D * D);
  cvt(stream, We, we16, (size_t)D * DE);
  cvt(stream, W1, w116, (size_t)Lc * D * 2 * D);
  cvt(stream, W2, w216, (size_t)Lc * D * D);
  cvt(stream, Wqkv, wqkv16, (size_t)3 * D * D);
  cvt(stream, Wo, wo16, (size_t)D * D);

  // nodes = node_features @ Wn^T + bn
  run_gemm_nk(stream, nf16, D, wn16, D, bn, nullptr, 0, nodes, D, Nn, D, D);
  // edge_proj = edge_features @ We^T + be  (computed once, reused each layer)
  run_gemm_nk(stream, ef16, DE, we16, DE, be, nullptr, 0, eproj, D, E, D, DE);

  for (int l = 0; l < Lc; ++l) {
    hipMemsetAsync(agg, 0, (size_t)Nn * D * sizeof(float), stream);
    edge_msg_kernel<<<E, 192, 0, stream>>>(nodes, eproj, src, dst, agg, E);

    // combined @ W1^T split into two GEMMs over the column halves of W1[l]
    const f16* W1l = w116 + (size_t)l * D * 2 * D;
    cvt(stream, nodes, x16, (size_t)Nn * D);
    run_gemm_nk(stream, x16, D, W1l, 2 * D, b1 + (size_t)l * D, nullptr, 0,
                h1, D, Nn, D, D);
    cvt(stream, agg, agg16, (size_t)Nn * D);
    run_gemm_nk(stream, agg16, D, W1l + D, 2 * D, nullptr, h1, D,
                h1, D, Nn, D, D);

    ln_relu_kernel<<<Nn, 256, 0, stream>>>(h1, ln_g + (size_t)l * D,
                                           ln_b + (size_t)l * D, h216);

    // nodes = h2 @ W2^T + b2 + nodes  (in-place residual: each elem owned by one lane)
    run_gemm_nk(stream, h216, D, w216 + (size_t)l * D * D, D, b2 + (size_t)l * D,
                nodes, D, nodes, D, Nn, D, D);
  }

  // qkv = nodes @ Wqkv^T + bqkv, then stage qkv as f16 for attention GEMMs
  cvt(stream, nodes, x16, (size_t)Nn * D);
  run_gemm_nk(stream, x16, D, wqkv16, D, bqkv, nullptr, 0, qkv, 3 * D, Nn, 3 * D, D);
  cvt(stream, qkv, qkv16, (size_t)Nn * 3 * D);

  const float alpha = 0.10206207261596575f;  // 96^-0.5
  for (int h = 0; h < H; ++h) {
    // scores = alpha * Qh @ Kh^T   (strided head views into qkv16)
    run_gemm_nk(stream, qkv16 + h * DH, 3 * D, qkv16 + D + h * DH, 3 * D,
                nullptr, nullptr, 0, scores, Nn, Nn, Nn, DH, alpha);
    softmax4096_f16_kernel<<<Nn, 256, 0, stream>>>(scores, p16);
    // attn[:, h*DH:(h+1)*DH] = P @ Vh   (B is K-major here)
    run_gemm_kn(stream, p16, Nn, qkv16 + 2 * D + h * DH, 3 * D,
                nullptr, nullptr, 0, attn + h * DH, D, Nn, DH, Nn);
  }

  // out = attn @ Wo^T + bo + nodes
  cvt(stream, attn, x16, (size_t)Nn * D);
  run_gemm_nk(stream, x16, D, wo16, D, bo, nodes, D, out, D, Nn, D, D);
}
